// LocalAttentionTransformerBaseline_48223892799523
// MI455X (gfx1250) — matrix operand order repaired and checked by reading, not compiled, not run on hardware
//
#include <hip/hip_runtime.h>
#include <math.h>

// ---------------------------------------------------------------------------
// Types for CDNA5 WMMA (gfx1250, wave32)
// ---------------------------------------------------------------------------
typedef __attribute__((ext_vector_type(16))) __bf16    v16bf;
typedef __attribute__((ext_vector_type(8)))  float     v8f;
typedef __attribute__((ext_vector_type(8)))  unsigned  v8u;
typedef unsigned short bf16raw;

// f32 -> bf16 bits (RNE)
__device__ __forceinline__ bf16raw f2bf_u16(float f) {
    unsigned u = __builtin_bit_cast(unsigned, f);
    u += 0x7FFFu + ((u >> 16) & 1u);
    return (bf16raw)(u >> 16);
}
// pack two f32 into bf16x2 dword (RNE + one v_perm_b32 merge)
__device__ __forceinline__ unsigned pack_bf16x2(float a, float b) {
    unsigned ua = __builtin_bit_cast(unsigned, a);
    unsigned ub = __builtin_bit_cast(unsigned, b);
    ua += 0x7FFFu + ((ua >> 16) & 1u);
    ub += 0x7FFFu + ((ub >> 16) & 1u);
    return __builtin_amdgcn_perm(ub, ua, 0x07060302u);  // {b[3],b[2],a[3],a[2]}
}

__device__ __forceinline__ float gelu_exact(float x) {
    return 0.5f * x * (1.0f + erff(x * 0.70710678118654752f));
}

// ---------------------------------------------------------------------------
// GEMM: C[M,N] = epilogue(A[M,K] @ W[K,N] + bias) (+ resid)
//   A  : bf16 row-major [M,K]  (pre-converted once by its producer)
//   Wp : dword-packed bf16 pairs along K: Wp[kp][n] = {W[2kp,n], W[2kp+1,n]}
// Invariants: M % 128 == 0, N % 32 == 0, K % 64 == 0  (ragged shapes pre-padded)
//   - block = 256 threads = 8 wave32s; block tile 128(M) x 32(N)
//   - each wave: one 16x32 tile = 2 accumulators, 2 x v_wmma_f32_16x16x32_bf16/step
//   - A fragment = 2 contiguous b128 loads/lane (bf16 row-major == fragment order)
//   - W 32x32 slab staged in double-buffered LDS (pure copy, no conversion)
//   - ACT: 0 = identity, 1 = exact GELU; OUT_BF16: C stored as bf16
// ---------------------------------------------------------------------------
template <int ACT, bool HAS_RESID, bool OUT_BF16>
__global__ __launch_bounds__(256)
void wmma_gemm_kernel(const bf16raw* __restrict__ Abf, const unsigned* __restrict__ Wp,
                      const float* __restrict__ bias, const float* __restrict__ resid,
                      void* __restrict__ Cout, int N, int K)
{
    const int tid  = threadIdx.x;
    const int wave = tid >> 5;
    const int lane = tid & 31;
    const int l15  = lane & 15;
    const int hl   = lane >> 4;            // lane half: 0 or 1

    const int n0 = blockIdx.x * 32;
    const int m0 = blockIdx.y * 128 + wave * 16;

    // double-buffered [khalf][col][pair]: lane's B fragment = 8 contiguous dwords
    __shared__ unsigned sWp[2][2][32][8];

    // staging decomposition (coalesced Wp reads: consecutive threads -> consecutive cols)
    const int sc = tid & 31;               // column within tile
    const int sj = (tid >> 5) & 7;         // K-pair index within half

    const bf16raw*  __restrict__ arow = Abf + (size_t)(m0 + l15) * K;
    const unsigned* __restrict__ wrow = Wp + (size_t)(n0 + sc) + (size_t)sj * N;

    v8f acc0 = {};
    v8f acc1 = {};

    for (int k0 = 0; k0 < K; k0 += 64) {
        // speculative prefetch of the next A slab (global_prefetch_b8)
        __builtin_prefetch(arow + k0 + 64, 0, 0);
        #pragma unroll
        for (int half = 0; half < 2; ++half) {
            const int kk = k0 + half * 32;
            // stage 32x32 W slab: pure copy of pre-packed dwords, 2 per thread
            #pragma unroll
            for (int hh = 0; hh < 2; ++hh)
                sWp[half][hh][sc][sj] = wrow[(size_t)(kk / 2 + hh * 8) * N];
            __syncthreads();   // single barrier: ds_load->wmma wait covers WAR

            // A fragment: bf16 row-major == fragment order -> 2 x b128 per lane
            const uint4 alo = *(const uint4*)(arow + kk + hl * 16);
            const uint4 ahi = *(const uint4*)(arow + kk + hl * 16 + 8);
            v8u au;
            au[0] = alo.x; au[1] = alo.y; au[2] = alo.z; au[3] = alo.w;
            au[4] = ahi.x; au[5] = ahi.y; au[6] = ahi.z; au[7] = ahi.w;
            const v16bf afrag = __builtin_bit_cast(v16bf, au);

            // B fragments: contiguous 32B LDS reads per lane
            const v16bf bfrag0 = __builtin_bit_cast(v16bf, *(const v8u*)sWp[half][hl][l15]);
            const v16bf bfrag1 = __builtin_bit_cast(v16bf, *(const v8u*)sWp[half][hl][l15 + 16]);

            acc0 = __builtin_amdgcn_wmma_f32_16x16x32_bf16(
                       false, afrag, false, bfrag0, (short)0, acc0, false, false);
            acc1 = __builtin_amdgcn_wmma_f32_16x16x32_bf16(
                       false, afrag, false, bfrag1, (short)0, acc1, false, false);
        }
    }

    // epilogue: C/D layout -> VGPR r: M = m0 + r + 8*hl, N = n0 + l15 (+16)
    const int col0 = n0 + l15;
    const int col1 = col0 + 16;
    const float bv0 = bias[col0];
    const float bv1 = bias[col1];
    #pragma unroll
    for (int r = 0; r < 8; ++r) {
        const size_t rowc = (size_t)(m0 + hl * 8 + r);
        float v0 = acc0[r] + bv0;
        float v1 = acc1[r] + bv1;
        if constexpr (ACT == 1) { v0 = gelu_exact(v0); v1 = gelu_exact(v1); }
        if constexpr (HAS_RESID) {
            v0 += resid[rowc * N + col0];
            v1 += resid[rowc * N + col1];
        }
        if constexpr (OUT_BF16) {
            bf16raw* C = (bf16raw*)Cout;
            C[rowc * N + col0] = f2bf_u16(v0);
            C[rowc * N + col1] = f2bf_u16(v1);
        } else {
            float* C = (float*)Cout;
            C[rowc * N + col0] = v0;
            C[rowc * N + col1] = v1;
        }
    }
}

// ---------------------------------------------------------------------------
// LayerNorm over D=256: one wave32 per row, f32 in, bf16 out (feeds GEMMs only)
// ---------------------------------------------------------------------------
__global__ __launch_bounds__(256)
void layernorm_kernel(const float* __restrict__ x, const float* __restrict__ g,
                      const float* __restrict__ b, bf16raw* __restrict__ y, int rows)
{
    const int wave = threadIdx.x >> 5;
    const int lane = threadIdx.x & 31;
    const int row  = blockIdx.x * 8 + wave;
    if (row >= rows) return;

    const float* xr = x + (size_t)row * 256;
    float v[8];
    float s = 0.0f;
    #pragma unroll
    for (int i = 0; i < 8; ++i) { v[i] = xr[lane + i * 32]; s += v[i]; }
    #pragma unroll
    for (int off = 16; off; off >>= 1) s += __shfl_xor(s, off, 32);
    const float mu = s * (1.0f / 256.0f);

    float vs = 0.0f;
    #pragma unroll
    for (int i = 0; i < 8; ++i) { float d = v[i] - mu; vs += d * d; }
    #pragma unroll
    for (int off = 16; off; off >>= 1) vs += __shfl_xor(vs, off, 32);
    const float rstd = rsqrtf(vs * (1.0f / 256.0f) + 1e-5f);

    bf16raw* yr = y + (size_t)row * 256;
    #pragma unroll
    for (int i = 0; i < 8; ++i) {
        int c = lane + i * 32;
        yr[c] = f2bf_u16((v[i] - mu) * rstd * g[c] + b[c]);
    }
}

// ---------------------------------------------------------------------------
// Banded local attention, radius 4 (window 9), per-batch sequences of F=2048.
// One wave32 per (b, f, head); lane = head-dim index (HD = 32).
// qkv f32 [4096, 768] = q|k|v; output bf16 (feeds out-proj GEMM only)
// ---------------------------------------------------------------------------
__global__ __launch_bounds__(256)
void local_attention_kernel(const float* __restrict__ qkv, bf16raw* __restrict__ o)
{
    const int wave = threadIdx.x >> 5;
    const int lane = threadIdx.x & 31;
    const int gid  = blockIdx.x * 8 + wave;      // (token, head) flat: 4096*8
    const int hd   = gid & 7;                    // head
    const int t    = gid >> 3;                   // token 0..4095
    const int f    = t & 2047;                   // position within sequence
    const float scale = 0.1767766952966369f;     // 1/sqrt(32)

    const float q = qkv[(size_t)t * 768 + hd * 32 + lane];

    float sc[9];
    float mx = -INFINITY;
    #pragma unroll
    for (int j = 0; j < 9; ++j) {
        const int fk = f + j - 4;                // wave-uniform branch
        float s;
        if (fk < 0 || fk >= 2048) {
            s = -INFINITY;
        } else {
            const int tk = t + j - 4;
            float p = q * qkv[(size_t)tk * 768 + 256 + hd * 32 + lane];
            #pragma unroll
            for (int off = 16; off; off >>= 1) p += __shfl_xor(p, off, 32);
            s = p * scale;
        }
        sc[j] = s;
        mx = fmaxf(mx, s);
    }

    float den = 0.0f;
    #pragma unroll
    for (int j = 0; j < 9; ++j) {
        sc[j] = (sc[j] == -INFINITY) ? 0.0f : expf(sc[j] - mx);
        den += sc[j];
    }
    const float inv = 1.0f / den;

    float accv = 0.0f;
    #pragma unroll
    for (int j = 0; j < 9; ++j) {
        const int fk = f + j - 4;
        if (fk >= 0 && fk < 2048) {
            const int tk = t + j - 4;
            accv += sc[j] * qkv[(size_t)tk * 768 + 512 + hd * 32 + lane];
        }
    }
    o[(size_t)t * 256 + hd * 32 + lane] = f2bf_u16(accv * inv);
}

// ---------------------------------------------------------------------------
// Weight packing: f32 [R1,C1] -> dword-packed bf16 K-pairs [K2/2, N2], zero-pad
// ---------------------------------------------------------------------------
__global__ __launch_bounds__(256)
void pack_weights_kernel(const float* __restrict__ src, unsigned* __restrict__ dst,
                         int R1, int C1, int N2, int total)   // total = K2/2 * N2
{
    const int i = blockIdx.x * 256 + threadIdx.x;
    if (i >= total) return;
    const int kp = i / N2, n = i % N2;
    const int k0 = 2 * kp, k1 = 2 * kp + 1;
    const float w0 = (k0 < R1 && n < C1) ? src[(size_t)k0 * C1 + n] : 0.0f;
    const float w1 = (k1 < R1 && n < C1) ? src[(size_t)k1 * C1 + n] : 0.0f;
    dst[i] = pack_bf16x2(w0, w1);
}

// x pad + bf16 convert: [T,51] f32 -> [T,64] bf16
__global__ __launch_bounds__(256)
void pad2d_bf16_kernel(const float* __restrict__ src, bf16raw* __restrict__ dst,
                       int C1, int C2, int total)
{
    const int i = blockIdx.x * 256 + threadIdx.x;
    if (i >= total) return;
    const int r = i / C2, c = i % C2;
    dst[i] = (c < C1) ? f2bf_u16(src[(size_t)r * C1 + c]) : (bf16raw)0;
}

__global__ __launch_bounds__(256)
void pad1d_kernel(const float* __restrict__ src, float* __restrict__ dst,
                  int C1, int C2)
{
    const int i = blockIdx.x * 256 + threadIdx.x;
    if (i >= C2) return;
    dst[i] = (i < C1) ? src[i] : 0.0f;
}

__global__ __launch_bounds__(256)
void unpad2d_kernel(const float* __restrict__ src, float* __restrict__ dst,
                    int C1, int C2, int total)         // src rows C1 wide -> dst C2 wide
{
    const int i = blockIdx.x * 256 + threadIdx.x;
    if (i >= total) return;
    const int r = i / C2, c = i % C2;
    dst[i] = src[(size_t)r * C1 + c];
}

// ---------------------------------------------------------------------------
// Host orchestration
// ---------------------------------------------------------------------------
static void gemm_plain(const bf16raw* A, const unsigned* Wp, const float* bias,
                       float* C, int M, int N, int K, hipStream_t s)
{
    dim3 grid(N / 32, M / 128);
    hipLaunchKernelGGL((wmma_gemm_kernel<0, false, false>), grid, dim3(256), 0, s,
                       A, Wp, bias, nullptr, (void*)C, N, K);
}
static void gemm_resid(const bf16raw* A, const unsigned* Wp, const float* bias,
                       const float* resid, float* C, int M, int N, int K, hipStream_t s)
{
    dim3 grid(N / 32, M / 128);
    hipLaunchKernelGGL((wmma_gemm_kernel<0, true, false>), grid, dim3(256), 0, s,
                       A, Wp, bias, resid, (void*)C, N, K);
}
static void gemm_gelu_bf16(const bf16raw* A, const unsigned* Wp, const float* bias,
                           bf16raw* C, int M, int N, int K, hipStream_t s)
{
    dim3 grid(N / 32, M / 128);
    hipLaunchKernelGGL((wmma_gemm_kernel<1, false, true>), grid, dim3(256), 0, s,
                       A, Wp, bias, nullptr, (void*)C, N, K);
}
static void pack_w(const float* src, unsigned* dst, int R1, int C1, int K2, int N2,
                   hipStream_t s)
{
    const int total = (K2 / 2) * N2;
    hipLaunchKernelGGL(pack_weights_kernel, dim3((total + 255) / 256), dim3(256), 0, s,
                       src, dst, R1, C1, N2, total);
}

extern "C" void kernel_launch(void* const* d_in, const int* in_sizes, int n_in,
                              void* d_out, int out_size, void* d_ws, size_t ws_size,
                              hipStream_t stream)
{
    constexpr int T    = 4096;   // B*F tokens
    constexpr int D    = 256;
    constexpr int DFF  = 1024;
    constexpr int KIN  = 51;     // JIN*CIN
    constexpr int KPAD = 64;
    constexpr int NOUT = 51;     // JOUT*COUT
    constexpr int NPAD = 64;
    constexpr int L    = 4;

    const float* x      = (const float*)d_in[0];
    const float* w_in   = (const float*)d_in[1];
    const float* b_in   = (const float*)d_in[2];
    const float* qkv_w  = (const float*)d_in[3];
    const float* qkv_b  = (const float*)d_in[4];
    const float* out_w  = (const float*)d_in[5];
    const float* out_b  = (const float*)d_in[6];
    const float* ln1_g  = (const float*)d_in[7];
    const float* ln1_b  = (const float*)d_in[8];
    const float* ln2_g  = (const float*)d_in[9];
    const float* ln2_b  = (const float*)d_in[10];
    const float* ff_w1  = (const float*)d_in[11];
    const float* ff_b1  = (const float*)d_in[12];
    const float* ff_w2  = (const float*)d_in[13];
    const float* ff_b2  = (const float*)d_in[14];
    const float* lnf_g  = (const float*)d_in[15];
    const float* lnf_b  = (const float*)d_in[16];
    const float* head_w = (const float*)d_in[17];
    const float* head_b = (const float*)d_in[18];
    float* out = (float*)d_out;

    // --- workspace carve-out (~36 MB) ---
    char* ws = (char*)d_ws;
    float*    h      = (float*)ws;               ws += (size_t)T * D * 4;       // 4 MB
    float*    qkvb   = (float*)ws;               ws += (size_t)T * 3 * D * 4;   // 12 MB
    float*    o_pad  = (float*)ws;               ws += (size_t)T * NPAD * 4;    // 1 MB
    float*    hb_pad = (float*)ws;               ws += NPAD * 4;
    bf16raw*  x_bf   = (bf16raw*)ws;             ws += (size_t)T * KPAD * 2;    // 0.5 MB
    bf16raw*  y_bf   = (bf16raw*)ws;             ws += (size_t)T * D * 2;       // 2 MB
    bf16raw*  ob_bf  = (bf16raw*)ws;             ws += (size_t)T * D * 2;       // 2 MB
    bf16raw*  ff_bf  = (bf16raw*)ws;             ws += (size_t)T * DFF * 2;     // 8 MB
    unsigned* wi_p   = (unsigned*)ws;            ws += (size_t)(KPAD/2) * D * 4;
    unsigned* qkv_p  = (unsigned*)ws;            ws += (size_t)L * (D/2) * 3*D * 4;
    unsigned* outw_p = (unsigned*)ws;            ws += (size_t)L * (D/2) * D * 4;
    unsigned* ff1_p  = (unsigned*)ws;            ws += (size_t)L * (D/2) * DFF * 4;
    unsigned* ff2_p  = (unsigned*)ws;            ws += (size_t)L * (DFF/2) * D * 4;
    unsigned* head_p = (unsigned*)ws;            ws += (size_t)(D/2) * NPAD * 4;

    const dim3 blk(256), lnGrid(T / 8), atGrid(T * 8 / 8);

    // --- one-time conversions: pad + bf16-pack all GEMM operands ---
    hipLaunchKernelGGL(pad2d_bf16_kernel, dim3((T * KPAD + 255) / 256), blk, 0, stream,
                       x, x_bf, KIN, KPAD, T * KPAD);
    hipLaunchKernelGGL(pad1d_kernel, dim3(1), blk, 0, stream, head_b, hb_pad, NOUT, NPAD);
    pack_w(w_in,   wi_p,   KIN,     D,    KPAD,    D,    stream);
    pack_w(qkv_w,  qkv_p,  L * D,   3*D,  L * D,   3*D,  stream);  // even K per layer: pairs never cross layers
    pack_w(out_w,  outw_p, L * D,   D,    L * D,   D,    stream);
    pack_w(ff_w1,  ff1_p,  L * D,   DFF,  L * D,   DFF,  stream);
    pack_w(ff_w2,  ff2_p,  L * DFF, D,    L * DFF, D,    stream);
    pack_w(head_w, head_p, D,       NOUT, D,       NPAD, stream);

    // input projection: [4096,64]bf16 @ [64,256] + b -> h (f32)
    gemm_plain(x_bf, wi_p, b_in, h, T, D, KPAD, stream);

    for (int l = 0; l < L; ++l) {
        const unsigned* qw  = qkv_p  + (size_t)l * (D/2) * 3 * D;
        const unsigned* ow  = outw_p + (size_t)l * (D/2) * D;
        const unsigned* w1  = ff1_p  + (size_t)l * (D/2) * DFF;
        const unsigned* w2  = ff2_p  + (size_t)l * (DFF/2) * D;
        const float* qb  = qkv_b + (size_t)l * 3 * D;
        const float* obb = out_b + (size_t)l * D;
        const float* g1  = ln1_g + (size_t)l * D;
        const float* bg1 = ln1_b + (size_t)l * D;
        const float* g2  = ln2_g + (size_t)l * D;
        const float* bg2 = ln2_b + (size_t)l * D;
        const float* fb1 = ff_b1 + (size_t)l * DFF;
        const float* fb2 = ff_b2 + (size_t)l * D;

        hipLaunchKernelGGL(layernorm_kernel, lnGrid, blk, 0, stream, h, g1, bg1, y_bf, T);
        gemm_plain(y_bf, qw, qb, qkvb, T, 3 * D, D, stream);
        hipLaunchKernelGGL(local_attention_kernel, atGrid, blk, 0, stream, qkvb, ob_bf);
        gemm_resid(ob_bf, ow, obb, h, h, T, D, D, stream);           // h += o@W + b
        hipLaunchKernelGGL(layernorm_kernel, lnGrid, blk, 0, stream, h, g2, bg2, y_bf, T);
        gemm_gelu_bf16(y_bf, w1, fb1, ff_bf, T, DFF, D, stream);     // GELU -> bf16
        gemm_resid(ff_bf, w2, fb2, h, h, T, D, DFF, stream);         // h += g@W2 + b
    }

    hipLaunchKernelGGL(layernorm_kernel, lnGrid, blk, 0, stream, h, lnf_g, lnf_b, y_bf, T);
    gemm_plain(y_bf, head_p, hb_pad, o_pad, T, NPAD, D, stream);
    hipLaunchKernelGGL(unpad2d_kernel, dim3((T * NOUT + 255) / 256), blk, 0, stream,
                       o_pad, out, NPAD, NOUT, T * NOUT);
}
